// ScaledDotProductAttention_21079699488995
// MI455X (gfx1250) — matrix operand order repaired and checked by reading, not compiled, not run on hardware
//
#include <hip/hip_runtime.h>
#include <hip/hip_bf16.h>

typedef __attribute__((ext_vector_type(16))) _Float16 v16h;
typedef __attribute__((ext_vector_type(8)))  _Float16 v8h;
typedef __attribute__((ext_vector_type(8)))  float    v8f;
typedef __attribute__((ext_vector_type(4)))  float    v4f;

#define N_EMB   1024
#define HEAD    64
#define SEQ_T   2048
#define BATCH   8
#define BT      (BATCH * SEQ_T)

// workspace layout (bytes)
#define WT_OFF  0                               // 3 * 64 * 1024 f16 = 384 KB
#define QH_OFF  (3 * HEAD * N_EMB * 2)
#define KH_OFF  (QH_OFF + BT * HEAD * 2)
#define VT_OFF  (KH_OFF + BT * HEAD * 2)        // total ~6.6 MB

static __device__ __forceinline__ v16h cat8(v8h lo, v8h hi) {
  v16h r;
#pragma unroll
  for (int i = 0; i < 8; ++i) { r[i] = lo[i]; r[i + 8] = hi[i]; }
  return r;
}

static __device__ __forceinline__ v16h ld_frag(const _Float16* p0, const _Float16* p1) {
  return cat8(*(const v8h*)p0, *(const v8h*)p1);
}

static __device__ __forceinline__ v8f wmma_f16(v16h a, v16h b, v8f c) {
  return __builtin_amdgcn_wmma_f32_16x16x32_f16(
      /*neg_a=*/false, a, /*neg_b=*/false, b,
      /*c_mod=*/(short)0, c, /*reuse_a=*/false, /*reuse_b=*/false);
}

// ---------------------------------------------------------------------------
// Kernel 1: transpose+convert Wq/Wk/Wv [E,H] f32 -> Wt[3][H][E] f16
// ---------------------------------------------------------------------------
__global__ void prep_weights_kernel(const float* __restrict__ Wq,
                                    const float* __restrict__ Wk,
                                    const float* __restrict__ Wv,
                                    _Float16* __restrict__ Wt) {
  int idx = blockIdx.x * 256 + threadIdx.x;     // exactly 3*64*1024 threads
  int m = idx >> 16;                            // / (64*1024)
  int r = idx & 65535;
  int h = r >> 10;                              // / 1024
  int e = r & 1023;
  const float* W = (m == 0) ? Wq : ((m == 1) ? Wk : Wv);
  Wt[idx] = (_Float16)W[e * HEAD + h];
}

// ---------------------------------------------------------------------------
// Kernel 2: projection. grid (BT/16, 3), block = 1 wave.
//   m=0: Qh[token][h] = 0.125 * X*Wq (f16, pre-scaled)
//   m=1: Kh[token][h]
//   m=2: Vt[b][h][t]   (transposed for B-fragments of P*V)
// ---------------------------------------------------------------------------
__global__ void __launch_bounds__(32)
proj_kernel(const float* __restrict__ X, const _Float16* __restrict__ Wt,
            _Float16* __restrict__ Qh, _Float16* __restrict__ Kh,
            _Float16* __restrict__ Vt) {
  const int lane = threadIdx.x & 31;
  const int ln = lane & 15, hi = lane >> 4;
  const int t0 = blockIdx.x * 16;
  const int m  = blockIdx.y;
  const _Float16* Wm = Wt + (size_t)m * HEAD * N_EMB;
  const float* xrow = X + (size_t)(t0 + ln) * N_EMB;

  v8f acc[4] = {};
  for (int c = 0; c < N_EMB / 32; ++c) {
    const int cb = c * 32 + hi * 8;
    // A fragment: 8+8 f32 -> f16
    v4f x0 = *(const v4f*)(xrow + cb);
    v4f x1 = *(const v4f*)(xrow + cb + 4);
    v4f x2 = *(const v4f*)(xrow + cb + 16);
    v4f x3 = *(const v4f*)(xrow + cb + 20);
    // all four B fragments first (one load clause, staggered waits)
    v16h bf[4];
#pragma unroll
    for (int nt = 0; nt < 4; ++nt) {
      const _Float16* wrow = Wm + (size_t)(nt * 16 + ln) * N_EMB + c * 32 + hi * 8;
      bf[nt] = ld_frag(wrow, wrow + 16);
    }
    v16h a;
#pragma unroll
    for (int i = 0; i < 4; ++i) {
      a[i]      = (_Float16)x0[i];
      a[i + 4]  = (_Float16)x1[i];
      a[i + 8]  = (_Float16)x2[i];
      a[i + 12] = (_Float16)x3[i];
    }
#pragma unroll
    for (int nt = 0; nt < 4; ++nt) acc[nt] = wmma_f16(a, bf[nt], acc[nt]);
  }

  // uniform branch on m, clean store loops
  if (m == 0) {
#pragma unroll
    for (int nt = 0; nt < 4; ++nt)
#pragma unroll
      for (int v = 0; v < 8; ++v)
        Qh[(size_t)(t0 + v + 8 * hi) * HEAD + nt * 16 + ln] =
            (_Float16)(acc[nt][v] * 0.125f);                 // 1/sqrt(64)
  } else if (m == 1) {
#pragma unroll
    for (int nt = 0; nt < 4; ++nt)
#pragma unroll
      for (int v = 0; v < 8; ++v)
        Kh[(size_t)(t0 + v + 8 * hi) * HEAD + nt * 16 + ln] = (_Float16)acc[nt][v];
  } else {
    const int bb = t0 >> 11;                    // batch (tile never spans batches)
    const int tbase = t0 & 2047;
    _Float16* Vbb = Vt + (size_t)bb * HEAD * SEQ_T;
#pragma unroll
    for (int nt = 0; nt < 4; ++nt)
#pragma unroll
      for (int v = 0; v < 8; ++v)
        Vbb[(size_t)(nt * 16 + ln) * SEQ_T + tbase + v + 8 * hi] = (_Float16)acc[nt][v];
  }
}

// ---------------------------------------------------------------------------
// Kernel 3: causal flash attention. grid (T/64, B), block = 4 waves,
// each wave owns one 16-query tile; key tiles of 32 with online softmax.
// ---------------------------------------------------------------------------
__global__ void __launch_bounds__(128)
attn_kernel(const _Float16* __restrict__ Qh, const _Float16* __restrict__ Kh,
            const _Float16* __restrict__ Vt, float* __restrict__ Out) {
  __shared__ __align__(16) _Float16 Plds[4][16 * 32];   // 1 KB per wave
  const int w    = threadIdx.x >> 5;
  const int lane = threadIdx.x & 31;
  const int ln = lane & 15, hi = lane >> 4;
  const int b   = blockIdx.y;
  const int qt0 = (blockIdx.x * 4 + w) * 16;

  const _Float16* Qb = Qh + (size_t)b * SEQ_T * HEAD;
  const _Float16* Kb = Kh + (size_t)b * SEQ_T * HEAD;
  const _Float16* Vb = Vt + (size_t)b * HEAD * SEQ_T;

  // resident Q fragments (16 x 64 = two 16x32 A-frags)
  v16h qa[2];
#pragma unroll
  for (int c = 0; c < 2; ++c) {
    const _Float16* p = Qb + (size_t)(qt0 + ln) * HEAD + c * 32 + hi * 8;
    qa[c] = ld_frag(p, p + 16);
  }

  v8f o[4] = {};
  float mrow[8], lrow[8];
#pragma unroll
  for (int v = 0; v < 8; ++v) { mrow[v] = -1e30f; lrow[v] = 0.f; }
  const float L2E = 1.44269504088896340736f;

  for (int kt0 = 0; kt0 <= qt0; kt0 += 32) {
    if (kt0 + 32 <= qt0) {
      __builtin_prefetch(Kb + (size_t)(kt0 + 32 + ln) * HEAD, 0, 0);
      __builtin_prefetch(Vb + (size_t)ln * SEQ_T + kt0 + 32, 0, 0);
    }

    // ---- K fragments: 8 x b128 in one batch ----
    const _Float16* kbase = Kb + (size_t)(kt0 + ln) * HEAD + hi * 8;
    v16h kf[2][2];                       // [tile t][k-chunk c]
#pragma unroll
    for (int t = 0; t < 2; ++t)
#pragma unroll
      for (int c = 0; c < 2; ++c) {
        const _Float16* p = kbase + (size_t)t * 16 * HEAD + c * 32;
        kf[t][c] = ld_frag(p, p + 16);
      }

    // S = Q K^T : two 16x16 tiles (keys kt0..+15, kt0+16..+31)
    v8f s0 = {}, s1 = {};
    s0 = wmma_f16(qa[0], kf[0][0], s0);
    s0 = wmma_f16(qa[1], kf[0][1], s0);
    s1 = wmma_f16(qa[0], kf[1][0], s1);
    s1 = wmma_f16(qa[1], kf[1][1], s1);

    // ---- issue V fragment loads now; softmax VALU hides their latency ----
    v16h vf[4];
#pragma unroll
    for (int nt = 0; nt < 4; ++nt) {
      const _Float16* vp = Vb + (size_t)(nt * 16 + ln) * SEQ_T + kt0 + hi * 8;
      vf[nt] = ld_frag(vp, vp + 16);
    }

    // causal mask (branchless per lane; EXEC stays full for WMMA)
    if (kt0 + 31 > qt0) {
#pragma unroll
      for (int v = 0; v < 8; ++v) {
        int q  = qt0 + v + 8 * hi;
        int k0 = kt0 + ln;
        s0[v] = (k0 > q)      ? -1e30f : s0[v];
        s1[v] = (k0 + 16 > q) ? -1e30f : s1[v];
      }
    }

    // online softmax: per row (v, half-lane group)
#pragma unroll
    for (int v = 0; v < 8; ++v) {
      float mt = fmaxf(s0[v], s1[v]);
#pragma unroll
      for (int d = 8; d >= 1; d >>= 1) mt = fmaxf(mt, __shfl_xor(mt, d, 32));
      float mn = fmaxf(mrow[v], mt);
      float alpha = exp2f((mrow[v] - mn) * L2E);
      mrow[v] = mn;
      float p0 = exp2f((s0[v] - mn) * L2E);
      float p1 = exp2f((s1[v] - mn) * L2E);
      float rs = p0 + p1;
#pragma unroll
      for (int d = 8; d >= 1; d >>= 1) rs += __shfl_xor(rs, d, 32);
      lrow[v] = lrow[v] * alpha + rs;
#pragma unroll
      for (int nt = 0; nt < 4; ++nt) o[nt][v] *= alpha;
      // C-layout -> LDS P[16][32] (f16), for re-read in A-layout
      Plds[w][(v + 8 * hi) * 32 + ln]      = (_Float16)p0;
      Plds[w][(v + 8 * hi) * 32 + 16 + ln] = (_Float16)p1;
    }
    asm volatile("s_wait_dscnt 0" ::: "memory");  // in-wave LDS RAW fence

    // P as 16x32 A-fragment
    const _Float16* pp = &Plds[w][ln * 32 + hi * 8];
    v16h pa = ld_frag(pp, pp + 16);

    // O += P * V  (V transposed: rows = h, contiguous keys)
#pragma unroll
    for (int nt = 0; nt < 4; ++nt) o[nt] = wmma_f16(pa, vf[nt], o[nt]);
  }

  float* Ob = Out + (size_t)b * SEQ_T * HEAD;
#pragma unroll
  for (int nt = 0; nt < 4; ++nt)
#pragma unroll
    for (int v = 0; v < 8; ++v)
      Ob[(size_t)(qt0 + v + 8 * hi) * HEAD + nt * 16 + ln] = o[nt][v] / lrow[v];
}

// ---------------------------------------------------------------------------
extern "C" void kernel_launch(void* const* d_in, const int* in_sizes, int n_in,
                              void* d_out, int out_size, void* d_ws, size_t ws_size,
                              hipStream_t stream) {
  const float* X  = (const float*)d_in[0];
  const float* Wq = (const float*)d_in[1];
  const float* Wk = (const float*)d_in[2];
  const float* Wv = (const float*)d_in[3];
  char* ws = (char*)d_ws;
  _Float16* Wt = (_Float16*)(ws + WT_OFF);
  _Float16* Qh = (_Float16*)(ws + QH_OFF);
  _Float16* Kh = (_Float16*)(ws + KH_OFF);
  _Float16* Vt = (_Float16*)(ws + VT_OFF);

  prep_weights_kernel<<<dim3(3 * HEAD * N_EMB / 256), dim3(256), 0, stream>>>(Wq, Wk, Wv, Wt);
  proj_kernel<<<dim3(BT / 16, 3), dim3(32), 0, stream>>>(X, Wt, Qh, Kh, Vt);
  attn_kernel<<<dim3(SEQ_T / 64, BATCH), dim3(128), 0, stream>>>(Qh, Kh, Vt, (float*)d_out);
}
